// DeformConvNet_88931592831261
// MI455X (gfx1250) — matrix operand order repaired and checked by reading, not compiled, not run on hardware
//
#include <hip/hip_runtime.h>

typedef __attribute__((ext_vector_type(16))) __bf16 v16bf;
typedef __attribute__((ext_vector_type(8)))  float  v8f;
typedef __attribute__((ext_vector_type(4)))  float  v4f;

// ---------------------------------------------------------------------------
// Weight pre-pack: OIHW f32 -> bf16 arranged in the exact wave32 WMMA A-matrix
// VGPR layout, so the conv kernel loads 16 bf16 (32 bytes) contiguously per
// lane per K-step.  K-step order: tap-major, then ci-block (matches conv loop).
// dst[((mt*Ksteps + s)*32 + lane)*16 + i]
// ---------------------------------------------------------------------------
__global__ void __launch_bounds__(256)
pack_weights_kernel(const float* __restrict__ w, unsigned short* __restrict__ dst,
                    int Ci, int Co)
{
    int total = Co * Ci * 9;
    int tid = blockIdx.x * 256 + threadIdx.x;
    if (tid >= total) return;

    int i    = tid & 15;
    int lane = (tid >> 4) & 31;
    int rest = tid >> 9;
    int nCb    = Ci >> 5;
    int Ksteps = 9 * nCb;
    int s  = rest % Ksteps;
    int mt = rest / Ksteps;
    int tap = s / nCb;
    int cbi = s - tap * nCb;

    int half = lane >> 4;
    int m    = lane & 15;
    int co   = mt * 16 + m;
    // A (16x32 bf16) layout: lanes 0-15 -> K = 0..7,16..23 ; lanes 16-31 -> K = 8..15,24..31
    int kA = (i < 8) ? (8 * half + i) : (8 * half + i + 8);
    int ci = cbi * 32 + kA;

    float v = w[(co * Ci + ci) * 9 + tap];
    unsigned u = __builtin_bit_cast(unsigned, v);
    dst[tid] = (unsigned short)((u + 0x7FFFu + ((u >> 16) & 1u)) >> 16); // f32->bf16 RNE
}

// ---------------------------------------------------------------------------
// Implicit-GEMM 3x3 conv, padding=1.  Compile-time CI/H/W/stride so all strided
// im2col offsets are 24-bit instruction immediates.  One wave32 computes a
// 32(Co) x 16(pixels) tile: each B gather feeds TWO v_wmma_f32_16x16x32_bf16,
// halving the dominant L2 gather traffic.  Branch-free bias+ReLU epilogue.
// ---------------------------------------------------------------------------
template <int CI, int H, int W, int S, bool HAS_BIAS, bool RELU>
__global__ void __launch_bounds__(256)
conv3x3_wmma(const float* __restrict__ in, const unsigned short* __restrict__ pw,
             const float* __restrict__ bias, float* __restrict__ out,
             int Co, int Mtiles2, int Ntiles, int totalWaves)
{
    constexpr int HW   = H * W;
    constexpr int Ho   = (H - 1) / S + 1;
    constexpr int Wo   = (W - 1) / S + 1;
    constexpr int HoWo = Ho * Wo;
    constexpr int nCb  = CI / 32;
    constexpr int Ksteps = 9 * nCb;

    int wave = blockIdx.x * 8 + (threadIdx.x >> 5);
    if (wave >= totalWaves) return;            // wave-uniform: EXEC stays all-1s for WMMA
    int lane = threadIdx.x & 31;
    int half = lane >> 4;
    int l16  = lane & 15;

    int mt2  = wave % Mtiles2;                 // pair-of-M-tiles index
    int rest = wave / Mtiles2;
    int nt   = rest % Ntiles;
    int b    = rest / Ntiles;

    int pix0 = nt << 4;
    int oh   = pix0 / Wo;
    int ow0  = pix0 - oh * Wo;                 // Wo multiple of 16 -> tile stays in one row
    int owB  = ow0 + l16;                      // B-matrix column (pixel) for this lane
    int co0  = mt2 * 32;

    // A: packed bf16, 32 contiguous bytes per lane per K-step; second M-tile at
    // a compile-time offset of Ksteps*512 elements.
    const unsigned short* pA = pw + ((size_t)(2 * mt2) * Ksteps * 512) + lane * 16;

    const float* inBase = in + (size_t)(b * CI) * HW + half * 16 * HW;

    v8f acc0 = {};
    v8f acc1 = {};

    for (int tap = 0; tap < 9; ++tap) {
        int kh = tap / 3, kw = tap - 3 * kh;
        int ih = oh  * S - 1 + kh;             // uniform across wave
        int iw = owB * S - 1 + kw;             // per-lane
        bool ok = (ih >= 0) & (ih < H) & (iw >= 0) & (iw < W);
        int ihc = ih < 0 ? 0 : (ih >= H ? H - 1 : ih);   // clamp -> always-valid address
        int iwc = iw < 0 ? 0 : (iw >= W ? W - 1 : iw);
        const float* bPtr = inBase + ihc * W + iwc;

#pragma unroll
        for (int cbi = 0; cbi < nCb; ++cbi) {
            v16bf av0 = *(const v16bf*)pA;                         // 2x global_load_b128
            v16bf av1 = *(const v16bf*)(pA + Ksteps * 512);        // imm offset
            pA += 512;
            v16bf bv;
#pragma unroll
            for (int i = 0; i < 16; ++i) {
                // B (32x16 bf16) layout: lane holds K = cbi*32 + 8*half + i (i<8), cbi*32 + 16 + 8*half + (i-8) (i>=8)
                float v = bPtr[(cbi * 32 + i + (i >= 8 ? 8 : 0) - 8 * half) * HW];   // unconditional, imm offset
                v = ok ? v : 0.0f;                     // v_cndmask, no branch
                bv[i] = (__bf16)v;
            }
            acc0 = __builtin_amdgcn_wmma_f32_16x16x32_bf16(
                       false, av0, false, bv, (short)0, acc0, false, false);
            acc1 = __builtin_amdgcn_wmma_f32_16x16x32_bf16(
                       false, av1, false, bv, (short)0, acc1, false, false);
        }
    }

    // Branch-free epilogue: D layout M = i + 8*half (+16 for second tile), N = l16.
    v4f bb0a = {}, bb0b = {}, bb1a = {}, bb1b = {};
    if constexpr (HAS_BIAS) {
        const float* bp = bias + co0 + 8 * half;
        bb0a = *(const v4f*)(bp);
        bb0b = *(const v4f*)(bp + 4);
        bb1a = *(const v4f*)(bp + 16);
        bb1b = *(const v4f*)(bp + 20);
    }

    float* oBase = out + ((size_t)b * Co + co0 + 8 * half) * (size_t)HoWo
                       + (size_t)oh * Wo + ow0 + l16;
#pragma unroll
    for (int i = 0; i < 8; ++i) {
        float v0 = acc0[i];
        float v1 = acc1[i];
        if constexpr (HAS_BIAS) {
            v0 += (i < 4) ? bb0a[i & 3] : bb0b[i & 3];
            v1 += (i < 4) ? bb1a[i & 3] : bb1b[i & 3];
        }
        if constexpr (RELU) {
            v0 = fmaxf(v0, 0.0f);
            v1 = fmaxf(v1, 0.0f);
        }
        oBase[i * HoWo]        = v0;           // imm-offset stores
        oBase[(16 + i) * HoWo] = v1;
    }
}

// ---------------------------------------------------------------------------
// Deformable bilinear sampling with the reference's *raw reshape* offset
// scrambling.
// ---------------------------------------------------------------------------
__global__ void __launch_bounds__(256)
deform_kernel(const float* __restrict__ x, const float* __restrict__ off,
              float* __restrict__ out, int BC, int C, int H, int W)
{
    long long total = (long long)BC * H * W;
    long long idx = (long long)blockIdx.x * blockDim.x + threadIdx.x;
    if (idx >= total) return;

    int w  = (int)(idx % W);
    long long t1 = idx / W;
    int h  = (int)(t1 % H);
    int bc = (int)(t1 / H);
    int c  = bc % C;
    int b  = bc / C;
    int HW = H * W;

    float offv[2];
#pragma unroll
    for (int t = 0; t < 2; ++t) {
        int pos = c * 2 * HW + 2 * (h * W + w) + t;   // flat pos within (2C,H,W) block
        int cp  = pos / HW;
        int rem = pos - cp * HW;
        int hp  = rem / W;
        int wp  = rem - hp * W;
        offv[t] = off[(((long long)b * (2 * C) + cp) * H + hp) * W + wp];
    }

    float r  = fminf(fmaxf(offv[0] + (float)h, 0.0f), (float)(H - 1));
    float cc = fminf(fmaxf(offv[1] + (float)w, 0.0f), (float)(W - 1));
    float r0f = floorf(r), r1f = ceilf(r), c0f = floorf(cc), c1f = ceilf(cc);
    int r0 = (int)r0f, r1 = (int)r1f, c0 = (int)c0f, c1 = (int)c1f;

    const float* xb = x + (long long)bc * HW;
    float v_lt = xb[r0 * W + c0];
    float v_rb = xb[r1 * W + c1];
    float v_lb = xb[r0 * W + c1];
    float v_rt = xb[r1 * W + c0];
    float fr = r - r0f, fc = cc - c0f;
    float vt = v_lt + (v_rt - v_lt) * fr;
    float vb = v_lb + (v_rb - v_lb) * fr;
    out[idx] = vt + (vb - vt) * fc;
}

// One block per (b,co): mean over HW pixels.
__global__ void __launch_bounds__(256)
mean_kernel(const float* __restrict__ in, float* __restrict__ out, int HW)
{
    __shared__ float sm[256];
    const float* p = in + (long long)blockIdx.x * HW;
    float s = 0.0f;
    for (int i = threadIdx.x; i < HW; i += 256) s += p[i];
    sm[threadIdx.x] = s;
    __syncthreads();
    for (int st = 128; st > 0; st >>= 1) {
        if ((int)threadIdx.x < st) sm[threadIdx.x] += sm[threadIdx.x + st];
        __syncthreads();
    }
    if (threadIdx.x == 0) out[blockIdx.x] = sm[0] / (float)HW;
}

// ---------------------------------------------------------------------------
// Host-side helpers
// ---------------------------------------------------------------------------
template <int CI, int H, int W, int S, bool HAS_BIAS, bool RELU>
static void launch_conv(const float* in, const unsigned short* pw, const float* bias,
                        float* out, int B, int Co, hipStream_t stream)
{
    constexpr int Ho = (H - 1) / S + 1;
    constexpr int Wo = (W - 1) / S + 1;
    int Mtiles2 = Co / 32;                    // every layer has Co % 32 == 0
    int Ntiles  = (Ho * Wo) / 16;
    int totalWaves = B * Mtiles2 * Ntiles;
    int blocks = (totalWaves + 7) / 8;
    conv3x3_wmma<CI, H, W, S, HAS_BIAS, RELU><<<blocks, 256, 0, stream>>>(
        in, pw, bias, out, Co, Mtiles2, Ntiles, totalWaves);
}

static void launch_pack(const float* w, unsigned short* dst, int Ci, int Co,
                        hipStream_t stream)
{
    int total = Co * Ci * 9;
    pack_weights_kernel<<<(total + 255) / 256, 256, 0, stream>>>(w, dst, Ci, Co);
}

static void launch_deform(const float* x, const float* off, float* out,
                          int BC, int C, int H, int W, hipStream_t stream)
{
    long long total = (long long)BC * H * W;
    int blocks = (int)((total + 255) / 256);
    deform_kernel<<<blocks, 256, 0, stream>>>(x, off, out, BC, C, H, W);
}

extern "C" void kernel_launch(void* const* d_in, const int* in_sizes, int n_in,
                              void* d_out, int out_size, void* d_ws, size_t ws_size,
                              hipStream_t stream)
{
    (void)in_sizes; (void)n_in; (void)out_size; (void)ws_size;
    const float* x   = (const float*)d_in[0];
    const float* w1  = (const float*)d_in[1];
    const float* b1  = (const float*)d_in[2];
    const float* ow1 = (const float*)d_in[3];
    const float* w2  = (const float*)d_in[4];
    const float* b2  = (const float*)d_in[5];
    const float* ow2 = (const float*)d_in[6];
    const float* w3  = (const float*)d_in[7];
    const float* b3  = (const float*)d_in[8];
    const float* ow3 = (const float*)d_in[9];
    const float* w4  = (const float*)d_in[10];
    const float* b4  = (const float*)d_in[11];
    float* out = (float*)d_out;

    char* ws = (char*)d_ws;
    float* bufA = (float*)(ws);                                         // offsets, max 268 MB
    float* bufB = (float*)(ws + (size_t)268435456);                     // conv outs, max 134 MB
    float* bufC = (float*)(ws + (size_t)268435456 + (size_t)134217728); // deformed, max 134 MB
    unsigned short* wp = (unsigned short*)(ws + (size_t)536870912);     // packed bf16 weights

    unsigned short* pw1  = wp + 0;        // 32*32*9   = 9216
    unsigned short* pow1 = wp + 9216;     // 64*32*9   = 18432
    unsigned short* pw2  = wp + 27648;    // 64*32*9   = 18432
    unsigned short* pow2 = wp + 46080;    // 128*64*9  = 73728
    unsigned short* pw3  = wp + 119808;   // 32*64*9   = 18432
    unsigned short* pow3 = wp + 138240;   // 64*32*9   = 18432
    unsigned short* pw4  = wp + 156672;   // 32*32*9   = 9216

    // Pre-pack all weights to WMMA-layout bf16 (tiny: 332 KB total).
    launch_pack(w1,  pw1,  32, 32,  stream);
    launch_pack(ow1, pow1, 32, 64,  stream);
    launch_pack(w2,  pw2,  32, 64,  stream);
    launch_pack(ow2, pow2, 64, 128, stream);
    launch_pack(w3,  pw3,  64, 32,  stream);
    launch_pack(ow3, pow3, 32, 64,  stream);
    launch_pack(w4,  pw4,  32, 32,  stream);

    // Stage 1: 256x256, C=32
    launch_conv<32, 256, 256, 1, true,  true >(x,    pw1,  b1,      bufB, 16, 32, stream); // h1
    launch_conv<32, 256, 256, 1, false, false>(bufB, pow1, nullptr, bufA, 16, 64, stream); // off1
    launch_deform(bufB, bufA, bufC, 16 * 32, 32, 256, 256, stream);                        // d1

    // Stage 2: stride-2 conv -> 128x128, C=64
    launch_conv<32, 256, 256, 2, true,  true >(bufC, pw2,  b2,      bufB, 16, 64,  stream); // h2
    launch_conv<64, 128, 128, 1, false, false>(bufB, pow2, nullptr, bufA, 16, 128, stream); // off2
    launch_deform(bufB, bufA, bufC, 16 * 64, 64, 128, 128, stream);                         // d2

    // Stage 3: stride-2 conv -> 64x64, C=32
    launch_conv<64, 128, 128, 2, true,  true >(bufC, pw3,  b3,      bufB, 16, 32, stream);  // h3
    launch_conv<32, 64, 64, 1, false, false>(bufB, pow3, nullptr, bufA, 16, 64, stream);    // off3
    launch_deform(bufB, bufA, bufC, 16 * 32, 32, 64, 64, stream);                           // d3

    // Stage 4: stride-2 conv -> 32x32, then global mean
    launch_conv<32, 64, 64, 2, true, true>(bufC, pw4, b4, bufA, 16, 32, stream);            // h4
    mean_kernel<<<16 * 32, 256, 0, stream>>>(bufA, out, 32 * 32);
}
